// RPN_24455543783976
// MI455X (gfx1250) — matrix operand order repaired and checked by reading, not compiled, not run on hardware
//
#include <hip/hip_runtime.h>
#include <math.h>

// ---------------------------------------------------------------------------
// RPN for Faster-RCNN on gfx1250 (MI455X).
// Dominant cost: 3x3 conv (512->512, 8x64x64) = 154.6 GFLOP -> implicit-GEMM
// on v_wmma_f32_16x16x32_f16 (wave32).  Conv kernel: 8 waves/block, each wave
// holds 4 WMMA accumulators (64 Cout x 16 px), double-buffered LDS B-tiles
// (1 barrier / K-step, 4 WMMAs per wave per barrier).
// ---------------------------------------------------------------------------

typedef __attribute__((ext_vector_type(16))) _Float16 v16h;
typedef __attribute__((ext_vector_type(8)))  _Float16 v8h;
typedef __attribute__((ext_vector_type(8)))  float    v8f;

#define N_IMG 8
#define CIN   512
#define COUT  512
#define HH    64
#define WW    64
#define HP    66
#define WP    66
#define KTOT  4608              /* 9*512 */
#define PIX   4096              /* 64*64 */
#define NA    36864             /* PIX*9 */
#define NSORT 65536
#define NPRE  2000
#define NPOST 300
#define KSTEPS 144              /* KTOT/32 */

// d_out layout (floats), concatenated in reference return order
#define OFF_PARAMS 0u
#define OFF_SCORE  1179648u
#define OFF_ROIS   1769472u
#define OFF_RIDX   1779072u
#define OFF_ANCHOR 1781472u

// workspace layout (bytes, all 256-aligned)
#define WS_XPAD 0u                    /* 8*512*66*66 f16 = 35,684,352 */
#define WS_WR   35684352u             /* 512*4608 f16    =  4,718,592 */
#define WS_WL   40402944u             /* 64*512 f16      =     65,536 */
#define WS_MID  40468480u             /* 8*4096*512 f16  = 33,554,432 */
#define WS_O54  74022912u             /* 8*4096*64  f32  =  8,388,608 */
#define WS_FG   82411520u             /* 8*36864 f32     =  1,179,648 */
#define WS_ROI  83591168u             /* 8*36864*4 f32   =  4,718,592 */
#define WS_KEY  88309760u             /* 8*65536 f32     =  2,097,152 */
#define WS_IDX  90406912u             /* 8*65536 i32     =  2,097,152 */

// ---------------- WMMA fragment helpers (contiguous-K layout, same for A and B) ---------------
// A (16x32 f16): lane L holds row M=L&15; halves j -> K=(L<16?0:16)+j
//                (16 contiguous K values, matching B).
__device__ __forceinline__ v16h load_a_frag(const _Float16* row, int lane) {
  const int s = (lane >> 4) << 4;
  v8h lo = *(const v8h*)(row + s);
  v8h hi = *(const v8h*)(row + s + 8);
  v16h a;
#pragma unroll
  for (int j = 0; j < 8; ++j) { a[j] = lo[j]; a[8 + j] = hi[j]; }
  return a;
}
// B (32x16 f16): lane L holds col N=L&15; halves j -> K=(L<16?0:16)+j
// (16 contiguous K values -> caller passes the already-offset pointer).
__device__ __forceinline__ v16h load_b_frag16(const _Float16* p) {
  v8h lo = *(const v8h*)(p);
  v8h hi = *(const v8h*)(p + 8);
  v16h b;
#pragma unroll
  for (int j = 0; j < 8; ++j) { b[j] = lo[j]; b[8 + j] = hi[j]; }
  return b;
}

// ------------------------- data-prep kernels -------------------------------
__global__ void pad_x_kernel(const float* __restrict__ x, _Float16* __restrict__ xpad) {
  size_t gid = (size_t)blockIdx.x * blockDim.x + threadIdx.x;
  const size_t total = (size_t)N_IMG * CIN * HP * WP;
  if (gid >= total) return;
  int xx = (int)(gid % WP);
  size_t t = gid / WP;
  int yy = (int)(t % HP);
  size_t c = t / HP;                       // n*512+ci
  float v = 0.0f;
  if (yy >= 1 && yy <= HH && xx >= 1 && xx <= WW)
    v = x[(c * HH + (size_t)(yy - 1)) * WW + (xx - 1)];
  xpad[gid] = (_Float16)v;
}

// conv_w (O,I,3,3) -> wr[co][kpos*512+ci] so implicit-GEMM K axis is contiguous
__global__ void repack_w_kernel(const float* __restrict__ conv_w, _Float16* __restrict__ wr) {
  int gid = blockIdx.x * blockDim.x + threadIdx.x;
  if (gid >= COUT * KTOT) return;
  int k = gid % KTOT, co = gid / KTOT;
  int kpos = k / CIN, ci = k - kpos * CIN;
  wr[gid] = (_Float16)conv_w[((size_t)co * CIN + ci) * 9 + kpos];
}

// 1x1 head weights packed as [64][512]: co 0..35 params, 36..53 score, pad 0
__global__ void pack_wl_kernel(const float* __restrict__ params_w,
                               const float* __restrict__ score_w,
                               _Float16* __restrict__ wl) {
  int gid = blockIdx.x * blockDim.x + threadIdx.x;
  if (gid >= 64 * CIN) return;
  int k = gid & 511, co = gid >> 9;
  float v = 0.0f;
  if (co < 36) v = params_w[(size_t)co * CIN + k];
  else if (co < 54) v = score_w[(size_t)(co - 36) * CIN + k];
  wl[gid] = (_Float16)v;
}

// ------------------- 3x3 conv as implicit GEMM (WMMA) ----------------------
// grid.x = n(8) * pixel-tiles(256); block = 256 threads = 8 waves.
// Wave wv computes Cout [wv*64, wv*64+64) x 16 pixels via 4 accumulators.
// B tile (32 K x 16 px) double-buffered in LDS: 1 barrier per K-step.
__global__ __launch_bounds__(256) void conv3x3_wmma(
    const _Float16* __restrict__ xpad, const _Float16* __restrict__ wr,
    const float* __restrict__ conv_b, _Float16* __restrict__ mid) {
  __shared__ _Float16 lds[2 * 16 * 40];    // 2 buffers, padded row stride 40
  const int t = blockIdx.x;
  const int n = t >> 8;
  const int pt = t & 255;
  const int y = pt >> 2;
  const int x0 = (pt & 3) << 4;
  const int wv = threadIdx.x >> 5;
  const int lane = threadIdx.x & 31;

  v8f acc[4] = {};
  const _Float16* xp = xpad + (size_t)n * CIN * HP * WP + (size_t)y * WP + x0;
  const _Float16* wbase = wr + (size_t)(wv * 64 + (lane & 15)) * KTOT;
  const int klds = threadIdx.x >> 3;        // 0..31  (ci within chunk)
  const int plds = (threadIdx.x & 7) << 1;  // 0,2..14 (pixel pair)
  _Float16* myl = &lds[plds * 40 + klds];
  const _Float16* bfr = &lds[(lane & 15) * 40 + ((lane >> 4) << 4)];

  // stage K-step `it` (k = it*32) into buffer `buf`
  auto stage = [&](int it, int buf) {
    const int kpos = it >> 4;               // 0..8
    const int ky = kpos / 3, kx = kpos - ky * 3;
    const int cib = (it & 15) << 5;         // 0..480
    const _Float16* s = xp + (size_t)(cib + klds) * (HP * WP) + ky * WP + kx;
    myl[buf * 640]      = s[plds];
    myl[buf * 640 + 40] = s[plds + 1];
  };

  stage(0, 0);
  __syncthreads();
#pragma unroll 2
  for (int it = 0; it < KSTEPS; ++it) {
    const int cur = it & 1;
    if (it + 1 < KSTEPS) stage(it + 1, cur ^ 1);   // overlap with WMMAs
    v16h b = load_b_frag16(bfr + cur * 640);
    const _Float16* ap = wbase + it * 32;
#pragma unroll
    for (int ct = 0; ct < 4; ++ct) {
      v16h a = load_a_frag(ap + (size_t)ct * 16 * KTOT, lane);
      acc[ct] = __builtin_amdgcn_wmma_f32_16x16x32_f16(false, a, false, b,
                                                       (short)0, acc[ct], false, false);
    }
    __syncthreads();                        // readers done + next stage visible
  }

  // D: lane holds pixel N=lane&15, Cout M = r + (lane>=16?8:0); fuse bias+ReLU
  const int pix = y * WW + x0 + (lane & 15);
  _Float16* mrow = mid + ((size_t)n * PIX + pix) * CIN;
#pragma unroll
  for (int ct = 0; ct < 4; ++ct) {
    const int coh = wv * 64 + ct * 16 + ((lane >> 4) << 3);
    v8h o;
#pragma unroll
    for (int r = 0; r < 8; ++r) {
      float v = acc[ct][r] + conv_b[coh + r];
      o[r] = (_Float16)fmaxf(v, 0.0f);
    }
    *(v8h*)(mrow + coh) = o;                // pixel-major NHWC
  }
}

// -------------------- 1x1 head GEMM (WMMA): 64 x 512 x pixels --------------
__global__ __launch_bounds__(256) void gemm1x1_wmma(
    const _Float16* __restrict__ mid, const _Float16* __restrict__ wl,
    float* __restrict__ out54) {
  const int wv = threadIdx.x >> 5;
  const int lane = threadIdx.x & 31;
  const int pt = blockIdx.x * 8 + wv;                 // 0..2047 pixel tiles
  v8f acc[4] = {};
  const _Float16* mrow =
      mid + ((size_t)pt * 16 + (lane & 15)) * CIN + ((lane >> 4) << 4);
  for (int kb = 0; kb < CIN; kb += 32) {
    v16h b = load_b_frag16(mrow + kb);                // B = activations (N=pixel)
#pragma unroll
    for (int ct = 0; ct < 4; ++ct) {                  // A = weights (M=co)
      const _Float16* ap = wl + (size_t)(ct * 16 + (lane & 15)) * CIN + kb;
      v16h a = load_a_frag(ap, lane);
      acc[ct] = __builtin_amdgcn_wmma_f32_16x16x32_f16(false, a, false, b,
                                                       (short)0, acc[ct], false, false);
    }
  }
  float* orow = out54 + ((size_t)pt * 16 + (lane & 15)) * 64 + ((lane >> 4) << 3);
#pragma unroll
  for (int ct = 0; ct < 4; ++ct)
    *(v8f*)(orow + ct * 16) = acc[ct];
}

// ------------- bias + layout + softmax(fg = sigmoid(s1-s0)) ----------------
__global__ void head_kernel(const float* __restrict__ out54,
                            const float* __restrict__ params_b,
                            const float* __restrict__ score_b,
                            float* __restrict__ dout, float* __restrict__ fg) {
  int i = blockIdx.x * blockDim.x + threadIdx.x;
  if (i >= N_IMG * PIX) return;
  int n = i >> 12, p = i & 4095;
  const float* o = out54 + (size_t)i * 64;
  float* dp = dout + OFF_PARAMS + (size_t)n * 147456 + (size_t)p * 36;
#pragma unroll
  for (int j = 0; j < 36; ++j) dp[j] = o[j] + params_b[j];
  float s[18];
  float* ds = dout + OFF_SCORE + (size_t)n * 73728 + (size_t)p * 18;
#pragma unroll
  for (int j = 0; j < 18; ++j) { s[j] = o[36 + j] + score_b[j]; ds[j] = s[j]; }
  float* f = fg + (size_t)n * NA + (size_t)p * 9;
#pragma unroll
  for (int a = 0; a < 9; ++a)
    f[a] = 1.0f / (1.0f + expf(s[2 * a] - s[2 * a + 1]));   // softmax fg prob
}

// ------- anchors + loc2bbox + clip + min-size filter + sort keys -----------
__global__ void prep_kernel(const float* __restrict__ fg,
                            const int* __restrict__ img_h, const int* __restrict__ img_w,
                            float* __restrict__ roi, float* __restrict__ key,
                            int* __restrict__ idx, float* __restrict__ dout) {
  int gid = blockIdx.x * blockDim.x + threadIdx.x;
  if (gid >= N_IMG * NSORT) return;
  int n = gid >> 16, i = gid & (NSORT - 1);
  if (i >= NA) { key[gid] = -__builtin_inff(); idx[gid] = i; return; }
  int pix = i / 9, a = i - pix * 9;
  int y = pix >> 6, x = pix & 63;
  int r = a / 3, sc = a - r * 3;
  const float ratios[3] = {0.5f, 1.0f, 2.0f};
  const float scales[3] = {8.0f, 16.0f, 32.0f};
  float sr = sqrtf(ratios[r]);
  float ahs = 16.0f * scales[sc] * sr;
  float aws = 16.0f * scales[sc] / sr;
  float cy = y * 16.0f + 8.0f, cx = x * 16.0f + 8.0f;
  float ay1 = cy - 0.5f * ahs, ax1 = cx - 0.5f * aws;
  float ay2 = cy + 0.5f * ahs, ax2 = cx + 0.5f * aws;
  if (n == 0) {
    float* da = dout + OFF_ANCHOR + (size_t)i * 4;
    da[0] = ay1; da[1] = ax1; da[2] = ay2; da[3] = ax2;
  }
  const float* loc = dout + OFF_PARAMS + (size_t)n * 147456 + (size_t)i * 4;
  float dy = loc[0], dx = loc[1], dh = loc[2], dw = loc[3];
  float ah = ay2 - ay1, aw = ax2 - ax1;
  float ccy = dy * ah + (ay1 + 0.5f * ah);
  float ccx = dx * aw + (ax1 + 0.5f * aw);
  float h2 = 0.5f * expf(dh) * ah, w2 = 0.5f * expf(dw) * aw;
  float imh = (float)img_h[0], imw = (float)img_w[0];
  float y1 = fminf(fmaxf(ccy - h2, 0.0f), imh);
  float x1 = fminf(fmaxf(ccx - w2, 0.0f), imw);
  float y2 = fminf(fmaxf(ccy + h2, 0.0f), imh);
  float x2 = fminf(fmaxf(ccx + w2, 0.0f), imw);
  float* rr = roi + ((size_t)n * NA + i) * 4;
  rr[0] = y1; rr[1] = x1; rr[2] = y2; rr[3] = x2;
  key[gid] = ((y2 - y1) >= 16.0f && (x2 - x1) >= 16.0f)
                 ? fg[(size_t)n * NA + i] : -__builtin_inff();
  idx[gid] = i;
}

// --------------- per-image descending bitonic sort (65536) -----------------
__global__ __launch_bounds__(1024) void sort_kernel(float* __restrict__ key,
                                                    int* __restrict__ idx) {
  const int n = blockIdx.x;
  float* k = key + (size_t)n * NSORT;
  int* id = idx + (size_t)n * NSORT;
  for (unsigned sz = 2; sz <= NSORT; sz <<= 1) {
    for (unsigned j = sz >> 1; j > 0; j >>= 1) {
      for (unsigned t = threadIdx.x; t < NSORT / 2; t += blockDim.x) {
        unsigned i = ((t & ~(j - 1)) << 1) | (t & (j - 1));
        unsigned p = i | j;
        bool desc = ((i & sz) == 0);
        float ki = k[i], kp = k[p];
        if (desc ? (ki < kp) : (ki > kp)) {
          k[i] = kp; k[p] = ki;
          int ti = id[i]; id[i] = id[p]; id[p] = ti;
        }
      }
      __syncthreads();
    }
  }
}

// ----------------- per-image NMS (2000 boxes in LDS) + compaction ----------
__global__ __launch_bounds__(256) void nms_kernel(
    const float* __restrict__ roi, const float* __restrict__ key,
    const int* __restrict__ idx, float* __restrict__ dout) {
  __shared__ float4 bx[NPRE];
  __shared__ unsigned char fin[NPRE], supp[NPRE], keep[NPRE];
  __shared__ int ord[NPOST];
  __shared__ int cnt;
  const int n = blockIdx.x;
  const int tid = threadIdx.x;
  for (int t = tid; t < NPRE; t += 256) {
    int src = idx[(size_t)n * NSORT + t];
    float kk = key[(size_t)n * NSORT + t];
    float4 b = make_float4(0.f, 0.f, 0.f, 0.f);
    if (src < NA) {
      const float* rr = roi + ((size_t)n * NA + src) * 4;
      b = make_float4(rr[0], rr[1], rr[2], rr[3]);
    }
    bx[t] = b;
    fin[t] = (kk != -__builtin_inff()) ? 1 : 0;
    supp[t] = 0;
  }
  __syncthreads();
  for (int i = 0; i < NPRE; ++i) {
    bool keep_i = (supp[i] == 0);          // nms-keep (independent of finite)
    __syncthreads();                       // all read supp[i] before writes
    if (tid == 0) keep[i] = (keep_i && fin[i]) ? 1 : 0;
    if (keep_i) {
      float4 bi = bx[i];
      float ai = (bi.z - bi.x) * (bi.w - bi.y);
      for (int j = tid; j < NPRE; j += 256) {
        float4 bj = bx[j];
        float iy = fminf(bi.z, bj.z) - fmaxf(bi.x, bj.x);
        float ix = fminf(bi.w, bj.w) - fmaxf(bi.y, bj.y);
        float inter = fmaxf(iy, 0.f) * fmaxf(ix, 0.f);
        float aj = (bj.z - bj.x) * (bj.w - bj.y);
        if (inter / (ai + aj - inter + 1e-9f) > 0.7f) supp[j] = 1;
      }
    }
    __syncthreads();
  }
  if (tid == 0) {                           // stable order: kept-in-score-order
    int c = 0;
    for (int j = 0; j < NPRE && c < NPOST; ++j)
      if (keep[j]) ord[c++] = j;
    cnt = c;
  }
  __syncthreads();
  for (int s = tid; s < NPOST; s += 256) {
    float4 b = (s < cnt) ? bx[ord[s]] : make_float4(0.f, 0.f, 0.f, 0.f);
    float* d = dout + OFF_ROIS + ((size_t)n * NPOST + s) * 4;
    d[0] = b.x; d[1] = b.y; d[2] = b.z; d[3] = b.w;
    dout[OFF_RIDX + (size_t)n * NPOST + s] = (float)n;
  }
}

// --------------------------------- launch ----------------------------------
extern "C" void kernel_launch(void* const* d_in, const int* in_sizes, int n_in,
                              void* d_out, int out_size, void* d_ws, size_t ws_size,
                              hipStream_t stream) {
  const float* x        = (const float*)d_in[0];
  const float* conv_w   = (const float*)d_in[1];
  const float* conv_b   = (const float*)d_in[2];
  const float* score_w  = (const float*)d_in[3];
  const float* score_b  = (const float*)d_in[4];
  const float* params_w = (const float*)d_in[5];
  const float* params_b = (const float*)d_in[6];
  const int*   img_h    = (const int*)d_in[7];
  const int*   img_w    = (const int*)d_in[8];
  float* dout = (float*)d_out;
  char*  ws   = (char*)d_ws;

  _Float16* xpad = (_Float16*)(ws + WS_XPAD);
  _Float16* wr   = (_Float16*)(ws + WS_WR);
  _Float16* wl   = (_Float16*)(ws + WS_WL);
  _Float16* mid  = (_Float16*)(ws + WS_MID);
  float*    o54  = (float*)(ws + WS_O54);
  float*    fg   = (float*)(ws + WS_FG);
  float*    roi  = (float*)(ws + WS_ROI);
  float*    key  = (float*)(ws + WS_KEY);
  int*      idx  = (int*)(ws + WS_IDX);

  pad_x_kernel  <<<69696, 256, 0, stream>>>(x, xpad);
  repack_w_kernel<<<9216, 256, 0, stream>>>(conv_w, wr);
  pack_wl_kernel<<<128, 256, 0, stream>>>(params_w, score_w, wl);
  conv3x3_wmma  <<<2048, 256, 0, stream>>>(xpad, wr, conv_b, mid);
  gemm1x1_wmma  <<<256, 256, 0, stream>>>(mid, wl, o54);
  head_kernel   <<<128, 256, 0, stream>>>(o54, params_b, score_b, dout, fg);
  prep_kernel   <<<2048, 256, 0, stream>>>(fg, img_h, img_w, roi, key, idx, dout);
  sort_kernel   <<<8, 1024, 0, stream>>>(key, idx);
  nms_kernel    <<<8, 256, 0, stream>>>(roi, key, idx, dout);
}
